// Basic_RSNN_STDP_88845693485080
// MI455X (gfx1250) — 16-bit and 8-bit matrix operand order repaired and checked by reading, compiled, not run on hardware
//
#include <hip/hip_runtime.h>
#include <hip/hip_bf16.h>
#include <stdint.h>

typedef __attribute__((ext_vector_type(16))) _Float16 v16h;
typedef __attribute__((ext_vector_type(8)))  float    v8f;
typedef __attribute__((ext_vector_type(8)))  int      v8i;
typedef __attribute__((ext_vector_type(2)))  float    f2;

#define H_DIM   4096
#define N_INPUT 1024
#define N_OUT   256
#define T_STEPS 128

#if defined(__has_builtin)
#if __has_builtin(__builtin_amdgcn_wmma_f32_16x16x4_f32)
#define HAVE_WMMA_F32X4 1
#endif
#endif

__global__ void k_zero(float* __restrict__ p, int n) {
  int i = blockIdx.x * blockDim.x + threadIdx.x;
  if (i < n) p[i] = 0.0f;
}

__global__ void k_eff(const float* __restrict__ rec, const float* __restrict__ mask,
                      const float* __restrict__ bin, float* __restrict__ eff) {
  size_t i = (size_t)(blockIdx.x * blockDim.x + threadIdx.x) * 4;
  float4 r = *(const float4*)(rec + i);
  float4 m = *(const float4*)(mask + i);
  float4 b = *(const float4*)(bin + i);
  float4 e;
  e.x = r.x * m.x * b.x; e.y = r.y * m.y * b.y;
  e.z = r.z * m.z * b.z; e.w = r.w * m.w * b.w;
  *(float4*)(eff + i) = e;
}

#if HAVE_WMMA_F32X4
__global__ void k_ff(const float* __restrict__ x, const float* __restrict__ w,
                     float* __restrict__ ff) {
  int wid  = threadIdx.x >> 5;
  int lane = threadIdx.x & 31;
  int tile = blockIdx.x * 4 + wid;
  int tr = tile >> 8;
  int th = tile & 255;
  int m  = lane & 15;
  int hf = lane >> 4;
  int t0 = tr * 16, h0 = th * 16;
  const float* xrow = x + (size_t)(t0 + m) * N_INPUT + 2 * hf;
  const float* wrow = w + (size_t)(h0 + m) * N_INPUT + 2 * hf;
  v8f c = {};
#pragma unroll 8
  for (int k0 = 0; k0 < N_INPUT; k0 += 4) {
    f2 a = *(const f2*)(xrow + k0);
    f2 b = *(const f2*)(wrow + k0);
    c = __builtin_amdgcn_wmma_f32_16x16x4_f32(false, a, false, b,
                                              (short)0, c, false, false);
  }
#pragma unroll
  for (int v = 0; v < 8; ++v) {
    int t = t0 + v + hf * 8;
    ff[(size_t)t * H_DIM + h0 + m] = c[v];
  }
}
#else
__global__ void k_ff(const float* __restrict__ x, const float* __restrict__ w,
                     float* __restrict__ ff) {
  int wid  = threadIdx.x >> 5;
  int lane = threadIdx.x & 31;
  int tile = blockIdx.x * 4 + wid;
  int tr = tile >> 8;
  int th = tile & 255;
  int m  = lane & 15;
  int hf = lane >> 4;
  int t0 = tr * 16, h0 = th * 16;
  const float* xrow = x + (size_t)(t0 + m) * N_INPUT;
  const float* wrow = w + (size_t)(h0 + m) * N_INPUT;
  v8f c = {};
  for (int k0 = 0; k0 < N_INPUT; k0 += 32) {
    v16h a, b;
#pragma unroll
    for (int p = 0; p < 8; ++p) {
      int kp = hf * 16 + (p << 1);
      f2 xv = *(const f2*)(xrow + k0 + kp);
      a[2 * p]     = (_Float16)xv.x;
      a[2 * p + 1] = (_Float16)xv.y;
    }
    const float* wp = wrow + k0 + hf * 16;
#pragma unroll
    for (int q = 0; q < 16; ++q) b[q] = (_Float16)wp[q];
    c = __builtin_amdgcn_wmma_f32_16x16x32_f16(false, a, false, b,
                                               (short)0, c, false, false);
  }
#pragma unroll
  for (int v = 0; v < 8; ++v) {
    int t = t0 + v + hf * 8;
    ff[(size_t)t * H_DIM + h0 + m] = c[v];
  }
}
#endif

__global__ void k_step_hidden(const float* __restrict__ eff, const float* __restrict__ ff,
                              const float* __restrict__ tau, const float* __restrict__ w_out,
                              float* __restrict__ h_mem, const float* __restrict__ spk_in,
                              float* __restrict__ spk_out, const int* __restrict__ list,
                              const int* __restrict__ count_p,
                              unsigned char* __restrict__ spk_u8,
                              float* __restrict__ o_part, int t) {
  __shared__ int   llist[H_DIM];
  __shared__ float lpart[4][64];
  __shared__ float lsnew[64];
  int tid = threadIdx.x;
  int j0 = blockIdx.x * 64;
  int tj = tid & 63, ti = tid >> 6;
  int cnt = count_p[0];
  cnt = __builtin_amdgcn_readfirstlane(cnt);
  for (int k = tid; k < cnt; k += 256) llist[k] = list[k];
  __syncthreads();
  int n0 = (cnt * ti) >> 2;
  int n1 = (cnt * (ti + 1)) >> 2;
  float acc = 0.0f;
  for (int n = n0; n < n1; ++n) {
    int i = llist[n];
    int npf = (n + 4 < n1) ? (n + 4) : n;
    __builtin_prefetch(&eff[(size_t)llist[npf] * H_DIM + (j0 + tj)], 0, 1);
    acc += eff[(size_t)i * H_DIM + (j0 + tj)];
  }
  lpart[ti][tj] = acc;
  __syncthreads();
  if (ti == 0) {
    int j = j0 + tj;
    float sum = lpart[0][tj] + lpart[1][tj] + lpart[2][tj] + lpart[3][tj];
    float cur = ff[(size_t)t * H_DIM + j] + 0.1f * sum;
    float sp  = spk_in[j];
    float mem = h_mem[j] * tau[j] * (1.0f - sp) + cur;
    float sn  = (mem >= 0.5f) ? 1.0f : 0.0f;
    h_mem[j] = mem;
    spk_out[j] = sn;
    spk_u8[(size_t)t * H_DIM + j] = (unsigned char)sn;
    lsnew[tj] = sn;
  }
  __syncthreads();
  float oacc = 0.0f;
  const float* wrow = w_out + (size_t)tid * H_DIM + j0;
  for (int jj = 0; jj < 64; ++jj) {
    float s = lsnew[jj];
    if (s != 0.0f) oacc += wrow[jj];
  }
  o_part[blockIdx.x * 256 + tid] = oacc;
}

__global__ void k_step_mid(const float* __restrict__ o_part, float* __restrict__ o_mem,
                           float* __restrict__ o_spk, float* __restrict__ out,
                           const float* __restrict__ spk_cur, int* __restrict__ list,
                           int* __restrict__ count_p, int t) {
  __shared__ int lscan[256];
  int tid = threadIdx.x;
  float dot = 0.0f;
  for (int b = 0; b < 64; ++b) dot += o_part[b * 256 + tid];
  float om = o_mem[tid] * 0.96f * (1.0f - o_spk[tid]) + dot;
  float os = (om >= 0.5f) ? 1.0f : 0.0f;
  o_mem[tid] = om;
  o_spk[tid] = os;
  out[t * N_OUT + tid] = os;
  int base = 0;
  for (int c0 = 0; c0 < H_DIM; c0 += 256) {
    int p = (spk_cur[c0 + tid] != 0.0f) ? 1 : 0;
    lscan[tid] = p;
    __syncthreads();
    for (int off = 1; off < 256; off <<= 1) {
      int v = lscan[tid];
      int add = (tid >= off) ? lscan[tid - off] : 0;
      __syncthreads();
      lscan[tid] = v + add;
      __syncthreads();
    }
    int tot  = lscan[255];
    int excl = lscan[tid] - p;
    if (p) list[base + excl] = c0 + tid;
    base += tot;
    __syncthreads();
  }
  if (tid == 0) count_p[0] = base;
}

__global__ void k_prep_s(const unsigned char* __restrict__ spk_u8,
                         unsigned char* __restrict__ S_mat,
                         unsigned char* __restrict__ spkT, float* __restrict__ Acnt) {
  int i = blockIdx.x * blockDim.x + threadIdx.x;
  int s1 = 0, s2 = 0, a = 0;
  for (int t = 0; t < T_STEPS; ++t) {
    int s0 = spk_u8[(size_t)t * H_DIM + i];
    int S = s0 + s1 + s2;
    S_mat[(size_t)i * T_STEPS + t] = (unsigned char)S;
    spkT[(size_t)i * T_STEPS + t] = (unsigned char)s0;
    a += S;
    s2 = s1; s1 = s0;
  }
  Acnt[i] = (float)a;
}

__global__ void k_final(const unsigned char* __restrict__ S_mat,
                        const unsigned char* __restrict__ spkT,
                        const float* __restrict__ Acnt, const float* __restrict__ recurrent,
                        float* __restrict__ rec_out) {
  int wid  = threadIdx.x >> 5;
  int lane = threadIdx.x & 31;
  int tile = blockIdx.x * 4 + wid;
  int ti = tile >> 8, tj = tile & 255;
  int i0 = ti * 16, j0 = tj * 16;
  int m = lane & 15, hf = lane >> 4;
  const unsigned char* arow = S_mat + (size_t)(i0 + m) * T_STEPS;
  const unsigned char* brow = spkT  + (size_t)(j0 + m) * T_STEPS;
  v8i c = {};
#pragma unroll
  for (int kb = 0; kb < T_STEPS; kb += 64) {
    v8i A, B;
#pragma unroll
    for (int v = 0; v < 8; ++v) {
      int ka = ((v >> 1) << 4) + ((v & 1) << 2) + hf * 8;
      A[v] = *(const int*)(arow + kb + ka);
      int kbo = ((v >> 1) << 4) + ((v & 1) << 2) + hf * 8;
      B[v] = *(const int*)(brow + kb + kbo);
    }
    c = __builtin_amdgcn_wmma_i32_16x16x64_iu8(false, A, false, B, c, false, false);
  }
#pragma unroll
  for (int v = 0; v < 8; ++v) {
    int i = i0 + v + hf * 8;
    size_t idx = (size_t)i * H_DIM + j0 + m;
    float r = __builtin_nontemporal_load(recurrent + idx);
    float val = r + 3.84f - 0.02f * Acnt[i] + 0.02f * (float)c[v];
    __builtin_nontemporal_store(val, rec_out + idx);
  }
}

extern "C" void kernel_launch(void* const* d_in, const int* in_sizes, int n_in,
                              void* d_out, int out_size, void* d_ws, size_t ws_size,
                              hipStream_t stream) {
  (void)in_sizes; (void)n_in; (void)out_size; (void)ws_size;
  const float* x    = (const float*)d_in[0];
  const float* wfc1 = (const float*)d_in[1];
  const float* wout = (const float*)d_in[2];
  const float* rec  = (const float*)d_in[3];
  const float* tau  = (const float*)d_in[4];
  const float* mask = (const float*)d_in[5];
  const float* bin  = (const float*)d_in[6];

  float* out     = (float*)d_out;
  float* out_spk = out;
  float* rec_out = out + T_STEPS * N_OUT;

  float* wsf   = (float*)d_ws;
  float* ff    = wsf;
  float* h_mem = wsf + 524288;
  float* spk0  = wsf + 528384;
  float* spk1  = wsf + 532480;
  float* o_mem = wsf + 536576;
  float* o_spk = wsf + 536832;
  int*   count = (int*)(wsf + 537088);
  float* o_prt = wsf + 537344;
  float* Acnt  = wsf + 553728;
  int*   list  = (int*)(wsf + 557824);
  unsigned char* u8b  = (unsigned char*)d_ws + (size_t)561920 * 4;
  unsigned char* SPK  = u8b;
  unsigned char* Smat = u8b + 524288;
  unsigned char* SpkT = u8b + 1048576;

  k_zero<<<51, 256, 0, stream>>>(h_mem, 12801);
  k_eff<<<16384, 256, 0, stream>>>(rec, mask, bin, rec_out);
  k_ff<<<512, 128, 0, stream>>>(x, wfc1, ff);
  for (int t = 0; t < T_STEPS; ++t) {
    const float* si = (t & 1) ? spk1 : spk0;
    float*       so = (t & 1) ? spk0 : spk1;
    k_step_hidden<<<64, 256, 0, stream>>>(rec_out, ff, tau, wout, h_mem,
                                          si, so, list, count, SPK, o_prt, t);
    k_step_mid<<<1, 256, 0, stream>>>(o_prt, o_mem, o_spk, out_spk,
                                      so, list, count, t);
  }
  k_prep_s<<<16, 256, 0, stream>>>(SPK, Smat, SpkT, Acnt);
  k_final<<<16384, 128, 0, stream>>>(Smat, SpkT, Acnt, rec, rec_out);
}
